// GCNConvAdj_16801912062239
// MI455X (gfx1250) — matrix operand order repaired and checked by reading, not compiled, not run on hardware
//
#include <hip/hip_runtime.h>

// GCN: out = D^-1/2 (A+I) D^-1/2 (x W) + b      B=8, N=2048, F=256
// Strategy:
//   k1: dinv[b][n] = rsqrt(1 + sum_k adj[b][n][k])                (stream 128MB)
//   k2: Wt[n][k]   = bf16(W[k][n])                                 (128KB, B-frag friendly)
//   k3: hT[b][col][row] = bf16( dinv[row] * (x @ W)[row][col] )    (WMMA bf16, transposed store)
//   k4: out[b][row][col] = dinv[row]*( (adj_bf16 @ hT) + hT[row][col] ) + bias[col]
//       adj converted to bf16 on the fly; B tiles staged into LDS with
//       GLOBAL_LOAD_ASYNC_TO_LDS_B128 (ASYNCcnt) + double buffering;
//       v_wmma_f32_16x16x32_bf16 with f32 accumulators.

#define BATCH 8
#define NNODE 2048
#define FDIM  256

typedef __attribute__((ext_vector_type(16))) __bf16 v16bf;
typedef __attribute__((ext_vector_type(8)))  __bf16 v8bf;
typedef __attribute__((ext_vector_type(8)))  float  v8f;

// A-fragment (16x32 bf16, wave32) per ISA table:
//   lane<16 : row = m0+lane,    K-chunks [kk..kk+7]  and [kk+16..kk+23]
//   lane>=16: row = m0+lane-16, K-chunks [kk+8..+15] and [kk+24..+31]
// p points at this lane's chunk0 (8 f32, 32B-aligned); chunk1 is +16 floats.
__device__ __forceinline__ v16bf load_a_frag_f32(const float* __restrict__ p) {
    float4 f0 = *(const float4*)(p);
    float4 f1 = *(const float4*)(p + 4);
    float4 f2 = *(const float4*)(p + 16);
    float4 f3 = *(const float4*)(p + 20);
    v16bf a;
    a[0]  = (__bf16)f0.x; a[1]  = (__bf16)f0.y; a[2]  = (__bf16)f0.z; a[3]  = (__bf16)f0.w;
    a[4]  = (__bf16)f1.x; a[5]  = (__bf16)f1.y; a[6]  = (__bf16)f1.z; a[7]  = (__bf16)f1.w;
    a[8]  = (__bf16)f2.x; a[9]  = (__bf16)f2.y; a[10] = (__bf16)f2.z; a[11] = (__bf16)f2.w;
    a[12] = (__bf16)f3.x; a[13] = (__bf16)f3.y; a[14] = (__bf16)f3.z; a[15] = (__bf16)f3.w;
    return a;
}

// Async memory -> LDS copy of 16 bytes (no VGPR staging, tracked by ASYNCcnt).
// lds_off: byte address within LDS (generic shared ptr truncated to 32 bits,
// per ISA aperture rule addr[31:0] = LDS offset). gptr: global address.
__device__ __forceinline__ void async_copy_b128(unsigned lds_off, const void* gptr) {
    asm volatile("global_load_async_to_lds_b128 %0, %1, off"
                 :: "v"(lds_off), "v"((unsigned long long)(uintptr_t)gptr)
                 : "memory");
}

// ---------------- k1: degree -> dinv -------------------------------------
__global__ void gcn_deg_kernel(const float* __restrict__ adj, float* __restrict__ dinv) {
    const int row  = blockIdx.x * 8 + (threadIdx.x >> 5);   // [0, B*N)
    const int lane = threadIdx.x & 31;
    const float* r = adj + (size_t)row * NNODE;
    float s = 0.f;
    for (int i = lane * 4; i < NNODE; i += 32 * 4) {
        float4 v = *(const float4*)(r + i);
        s += v.x + v.y + v.z + v.w;
    }
    #pragma unroll
    for (int off = 16; off > 0; off >>= 1) s += __shfl_xor(s, off, 32);
    if (lane == 0) {
        float deg = s + 1.0f;                   // +1 from A+I
        dinv[row] = (deg > 0.f) ? rsqrtf(deg) : 0.f;
    }
}

// ---------------- k2: W -> Wt (bf16, transposed) --------------------------
__global__ void gcn_wt_kernel(const float* __restrict__ W, __bf16* __restrict__ Wt) {
    const int t = blockIdx.x * 256 + threadIdx.x;           // 65536 threads
    const int n = t >> 8, k = t & 255;
    Wt[(size_t)n * FDIM + k] = (__bf16)W[(size_t)k * FDIM + n];
}

// ---------------- k3: h' = dinv * (x @ W), stored transposed bf16 ---------
// grid(2,16,8): colBlock(128 cols), rowBlock(128 rows), batch. 8 waves/block.
__global__ void gcn_xw_kernel(const float* __restrict__ x, const __bf16* __restrict__ Wt,
                              const float* __restrict__ dinv, __bf16* __restrict__ hT) {
    const int lane = threadIdx.x & 31;
    const int wave = threadIdx.x >> 5;
    const int hi   = lane >> 4;                             // lane group 0/1
    const int b    = blockIdx.z;
    const int m0   = blockIdx.y * 128 + wave * 16;
    const int c0   = blockIdx.x * 128;

    const int rowA = m0 + (lane & 15);
    const float* xrow = x + ((size_t)(b * NNODE + rowA)) * FDIM;

    v8f acc[8] = {};
    for (int kk = 0; kk < FDIM; kk += 32) {
        v16bf a = load_a_frag_f32(xrow + kk + hi * 8);
        const int ks = kk + hi * 8;                         // B-frag K start
        #pragma unroll
        for (int t = 0; t < 8; ++t) {
            const int coln = c0 + t * 16 + (lane & 15);
            const uint4* bp = (const uint4*)(Wt + (size_t)coln * FDIM + ks);
            union { uint4 q[2]; v16bf v; } ub;
            ub.q[0] = bp[0]; ub.q[1] = bp[2];
            acc[t] = __builtin_amdgcn_wmma_f32_16x16x32_bf16(
                         false, a, false, ub.v, (short)0, acc[t], false, false);
        }
    }
    // epilogue: scale by dinv[row], pack bf16, store transposed (row-contiguous)
    const int row0 = m0 + (hi << 3);
    float4 d0 = *(const float4*)(dinv + b * NNODE + row0);
    float4 d1 = *(const float4*)(dinv + b * NNODE + row0 + 4);
    const float dv[8] = {d0.x, d0.y, d0.z, d0.w, d1.x, d1.y, d1.z, d1.w};
    #pragma unroll
    for (int t = 0; t < 8; ++t) {
        const int coln = c0 + t * 16 + (lane & 15);
        union { uint4 q; v8bf v; } o;
        #pragma unroll
        for (int r = 0; r < 8; ++r) o.v[r] = (__bf16)(acc[t][r] * dv[r]);
        *(uint4*)(hT + ((size_t)(b * FDIM + coln)) * NNODE + row0) = o.q;
    }
}

// ---------------- k4: out = dinv*(adj @ h') + dinv*h' + bias --------------
// grid(2,16,8): colBlock(128 cols), rowBlock(128 rows), batch. 8 waves/block,
// wave = 16-row subtile x 128 cols. B tiles (32K x 128 cols, 8KB) staged into
// a double-buffered LDS ring via async global->LDS b128 copies.
__global__ void gcn_spmm_kernel(const float* __restrict__ adj, const __bf16* __restrict__ hT,
                                const float* __restrict__ dinv, const float* __restrict__ bias,
                                float* __restrict__ out) {
    // per column: 64B of data + 16B pad (stride 80B keeps 16B alignment, breaks bank conflicts)
    __shared__ __align__(16) unsigned char ldsB[2][128 * 80];

    const int tid  = threadIdx.x;
    const int lane = tid & 31;
    const int wave = tid >> 5;
    const int hi   = lane >> 4;
    const int b    = blockIdx.z;
    const int m0   = blockIdx.y * 128 + wave * 16;
    const int c0   = blockIdx.x * 128;

    const int rowA = m0 + (lane & 15);
    const float* arow = adj + ((size_t)(b * NNODE + rowA)) * NNODE;

    // cooperative B staging: 512 x 16B chunks per K-step, 2 per thread
    const int colc0 = tid >> 2, seg0 = tid & 3;
    const int colc1 = colc0 + 64, seg1 = seg0;
    const __bf16* src0 = hT + ((size_t)(b * FDIM + c0 + colc0)) * NNODE + seg0 * 8;
    const __bf16* src1 = hT + ((size_t)(b * FDIM + c0 + colc1)) * NNODE + seg1 * 8;
    const unsigned ldsoff0 = (unsigned)(uintptr_t)&ldsB[0][colc0 * 80 + seg0 * 16];
    const unsigned ldsoff1 = (unsigned)(uintptr_t)&ldsB[0][colc1 * 80 + seg1 * 16];
    const unsigned bufstride = 128 * 80;

    // preload buffer 0 for kk = 0 (2 async ops outstanding per wave)
    async_copy_b128(ldsoff0, src0);
    async_copy_b128(ldsoff1, src1);

    v8f acc[8] = {};
    for (int kk = 0; kk < NNODE; kk += 32) {
        const unsigned cur = (kk >> 5) & 1;
        const unsigned nxt = cur ^ 1;
        const int kn = (kk + 32) & (NNODE - 1);   // wraps to 0 on last iter (dead prefetch)

        // all waves finished reading buf[nxt] (their previous 'cur') before we overwrite it
        __syncthreads();
        async_copy_b128(ldsoff0 + nxt * bufstride, src0 + kn);
        async_copy_b128(ldsoff1 + nxt * bufstride, src1 + kn);
        // wait for buf[cur]'s 2 async ops (in-order completion); next buffer's 2 stay in flight
        asm volatile("s_wait_asynccnt 0x2" ::: "memory");
        __syncthreads();                           // buf[cur] visible block-wide

        v16bf a = load_a_frag_f32(arow + kk + hi * 8);   // adj -> bf16 on the fly
        const unsigned char* bufc = &ldsB[cur][0];
        #pragma unroll
        for (int t = 0; t < 8; ++t) {
            const int coll = t * 16 + (lane & 15);
            const uint4* bp = (const uint4*)&bufc[coll * 80 + hi * 16];
            union { uint4 q[2]; v16bf v; } ub;
            ub.q[0] = bp[0]; ub.q[1] = bp[2];
            acc[t] = __builtin_amdgcn_wmma_f32_16x16x32_bf16(
                         false, a, false, ub.v, (short)0, acc[t], false, false);
        }
    }
    // epilogue: out = dinv[row]*(acc + h'[row][col]) + bias[col]
    const int row0 = m0 + (hi << 3);
    float4 d0 = *(const float4*)(dinv + b * NNODE + row0);
    float4 d1 = *(const float4*)(dinv + b * NNODE + row0 + 4);
    const float dv[8] = {d0.x, d0.y, d0.z, d0.w, d1.x, d1.y, d1.z, d1.w};
    #pragma unroll
    for (int t = 0; t < 8; ++t) {
        const int coll = c0 + t * 16 + (lane & 15);
        union { uint4 q; v8bf v; } hh;
        hh.q = *(const uint4*)(hT + ((size_t)(b * FDIM + coll)) * NNODE + row0);
        const float bs = bias[coll];
        #pragma unroll
        for (int r = 0; r < 8; ++r) {
            out[((size_t)(b * NNODE + row0 + r)) * FDIM + coll] =
                dv[r] * (acc[t][r] + (float)hh.v[r]) + bs;
        }
    }
}

extern "C" void kernel_launch(void* const* d_in, const int* in_sizes, int n_in,
                              void* d_out, int out_size, void* d_ws, size_t ws_size,
                              hipStream_t stream) {
    (void)in_sizes; (void)n_in; (void)out_size; (void)ws_size;
    const float* x    = (const float*)d_in[0];   // [8,2048,256]
    const float* adj  = (const float*)d_in[1];   // [8,2048,2048]
    const float* W    = (const float*)d_in[2];   // [256,256]
    const float* bias = (const float*)d_in[3];   // [256]
    float* out = (float*)d_out;                  // [8,2048,256]

    char* ws = (char*)d_ws;
    float*  dinv = (float*)ws;                              // 64 KB
    __bf16* Wt   = (__bf16*)(ws + 65536);                   // 128 KB
    __bf16* hT   = (__bf16*)(ws + 65536 + 131072);          // 8 MB  [B][col][row]

    gcn_deg_kernel<<<dim3(BATCH * NNODE / 8), dim3(256), 0, stream>>>(adj, dinv);
    gcn_wt_kernel <<<dim3(FDIM * FDIM / 256), dim3(256), 0, stream>>>(W, Wt);
    gcn_xw_kernel <<<dim3(2, 16, 8), dim3(256), 0, stream>>>(x, Wt, dinv, hT);
    gcn_spmm_kernel<<<dim3(2, 16, 8), dim3(256), 0, stream>>>(adj, hT, dinv, bias, out);
}
